// RotarySelfAttention_9148280340982
// MI455X (gfx1250) — compile-verified, matrix operand order checked by reading, not run on hardware
//
#include <hip/hip_runtime.h>
#include <hip/hip_bf16.h>

typedef __bf16 bf16_t;
typedef __bf16 v16bf __attribute__((ext_vector_type(16)));
typedef __bf16 v8bf  __attribute__((ext_vector_type(8)));
typedef float  v8f   __attribute__((ext_vector_type(8)));

#define T_SEQ  2048
#define C_DIM  1024
#define N_QKV  3072
#define H_N    16
#define D_H    64
#define B_N    4
#define M_ROWS 8192   // B*T

// ---- CDNA5 async global->LDS copy (ASYNCcnt-tracked), GVS addressing ----
__device__ __forceinline__ void async_ld16(unsigned int lds_addr, const void* sbase,
                                           unsigned int goff_bytes) {
    asm volatile("global_load_async_to_lds_b128 %0, %1, %2"
                 :
                 : "v"(lds_addr), "v"(goff_bytes), "s"(sbase)
                 : "memory");
}
#define ASYNC_WAIT_4() asm volatile("s_wait_asynccnt 0x4" ::: "memory")
#define ASYNC_WAIT_0() asm volatile("s_wait_asynccnt 0x0" ::: "memory")

// Load a 16-element bf16 A/B WMMA fragment slice.
// Caller passes p = row_base + (lane>=16 ? 8 : 0) [+ k0].
// Elements 0..7  <- p[0..7]   (K = kb .. kb+7)
// Elements 8..15 <- p[16..23] (K = kb+16 .. kb+23)
// Matches CDNA5 16-bit A/B layout: K(e,lane) = (e>=8?16:0)+(lane>=16?8:0)+(e%8).
__device__ __forceinline__ v16bf ld_frag(const bf16_t* p) {
    v8bf lo = *(const v8bf*)(p);
    v8bf hi = *(const v8bf*)(p + 16);
    return __builtin_shufflevector(lo, hi, 0,1,2,3,4,5,6,7,8,9,10,11,12,13,14,15);
}

__device__ __forceinline__ v8f wmma_bf16(v16bf a, v16bf b, v8f c) {
    return __builtin_amdgcn_wmma_f32_16x16x32_bf16(false, a, false, b, (short)0, c, false, false);
}

// Issue this thread's 4 async 16B copies for one 128x32 A-tile + 128x32 B-tile stage.
__device__ __forceinline__ void stage_async(unsigned int ldsA, unsigned int ldsB,
                                            const bf16_t* A, const bf16_t* Bt,
                                            int m0, int n0, int k0, int tid) {
#pragma unroll
    for (int i = 0; i < 2; ++i) {
        int chunk = tid * 2 + i;           // 0..511
        int row   = chunk >> 2;            // 0..127
        int col   = (chunk & 3) * 8;       // 0,8,16,24
        unsigned int ldso = (unsigned int)(row * 32 + col) * 2u;
        async_ld16(ldsA + ldso, A,  ((unsigned int)(m0 + row) * C_DIM + k0 + col) * 2u);
        async_ld16(ldsB + ldso, Bt, ((unsigned int)(n0 + row) * C_DIM + k0 + col) * 2u);
    }
}

// ---------------- stage 0: conversions ----------------

__global__ void f32_to_bf16_kernel(const float* __restrict__ in, bf16_t* __restrict__ out, int n) {
    int i = blockIdx.x * blockDim.x + threadIdx.x;
    if (i < n) out[i] = (bf16_t)in[i];
}

// out[n*K + k] = (bf16) in[k*N + n]  (W stored [K][N] row-major -> [N][K])
__global__ void transpose_bf16_kernel(const float* __restrict__ in, bf16_t* __restrict__ out, int K, int N) {
    int idx = blockIdx.x * blockDim.x + threadIdx.x;
    if (idx >= K * N) return;
    int n = idx / K;
    int k = idx - n * K;
    out[idx] = (bf16_t)in[(size_t)k * N + n];
}

// ---------------- stage 1: QKV GEMM + split ----------------
// C[m][n] = sum_k x[m][k] * Wt[n][k] + bias[n]; split into q,k ([BH][T][Dh]) and v ([BH][Dh][T]).
__global__ __launch_bounds__(256) void gemm_qkv_kernel(
    const bf16_t* __restrict__ A,   // [8192][1024]
    const bf16_t* __restrict__ Bt,  // [3072][1024]
    const float*  __restrict__ bias,// [3072]
    bf16_t* __restrict__ qbuf, bf16_t* __restrict__ kbuf, bf16_t* __restrict__ vbuf)
{
    __shared__ __attribute__((aligned(16))) bf16_t As[2][128 * 32];
    __shared__ __attribute__((aligned(16))) bf16_t Bs[2][128 * 32];
    const int tid  = threadIdx.x;
    const int lane = tid & 31, wv = tid >> 5;
    const int wr = wv & 3, wc = wv >> 2;          // wave tile: rows wr*32, cols wc*64
    const int half = lane >> 4, ln = lane & 15;
    const int m0 = blockIdx.y * 128;
    const int n0 = blockIdx.x * 128;
    const unsigned int ldsA0 = (unsigned int)(uintptr_t)&As[0][0];
    const unsigned int ldsB0 = (unsigned int)(uintptr_t)&Bs[0][0];
    const unsigned int bufBytes = 128u * 32u * 2u;

    v8f acc[2][4] = {};
    const int KT = C_DIM / 32;
    stage_async(ldsA0, ldsB0, A, Bt, m0, n0, 0, tid);     // prefetch tile 0
    for (int kt = 0; kt < KT; ++kt) {
        const int cur = kt & 1;
        if (kt + 1 < KT) {
            // next tile into the other buffer (its readers passed last iteration's barrier)
            stage_async(ldsA0 + (cur ^ 1) * bufBytes, ldsB0 + (cur ^ 1) * bufBytes,
                        A, Bt, m0, n0, (kt + 1) * 32, tid);
            ASYNC_WAIT_4();   // drain this wave's stage-kt copies (4 older of 8 outstanding)
        } else {
            ASYNC_WAIT_0();
        }
        __syncthreads();      // all waves drained their stage-kt copies -> tile visible

        v16bf af[2], bfg[4];
#pragma unroll
        for (int mi = 0; mi < 2; ++mi)
            af[mi] = ld_frag(&As[cur][(wr * 32 + mi * 16 + ln) * 32 + half * 8]);
#pragma unroll
        for (int ni = 0; ni < 4; ++ni)
            bfg[ni] = ld_frag(&Bs[cur][(wc * 64 + ni * 16 + ln) * 32 + half * 8]);
#pragma unroll
        for (int mi = 0; mi < 2; ++mi)
#pragma unroll
            for (int ni = 0; ni < 4; ++ni)
                acc[mi][ni] = wmma_bf16(af[mi], bfg[ni], acc[mi][ni]);
        __syncthreads();      // everyone done reading buffer `cur` before it is overwritten
    }

    // epilogue: bias + scatter to q/k/v
#pragma unroll
    for (int mi = 0; mi < 2; ++mi) {
#pragma unroll
        for (int ni = 0; ni < 4; ++ni) {
            const int gm = m0 + wr * 32 + mi * 16 + half * 8;
            const int gn = n0 + wc * 64 + ni * 16 + ln;
            const float bv = bias[gn];
            const int which = gn >> 10;            // 0=q 1=k 2=v
            const int hh    = (gn & 1023) >> 6;    // head
            const int d     = gn & 63;
#pragma unroll
            for (int r = 0; r < 8; ++r) {
                const int m = gm + r;
                const int b = m >> 11;             // / 2048
                const int t = m & 2047;
                const int bh = b * H_N + hh;
                const bf16_t val = (bf16_t)(acc[mi][ni][r] + bv);
                if (which == 0)      qbuf[((size_t)bh * T_SEQ + t) * D_H + d] = val;
                else if (which == 1) kbuf[((size_t)bh * T_SEQ + t) * D_H + d] = val;
                else                 vbuf[((size_t)bh * D_H + d) * T_SEQ + t] = val;
            }
        }
    }
}

// ---------------- stage 2: RoPE on q,k ----------------
__global__ void rope_kernel(bf16_t* __restrict__ qbuf, bf16_t* __restrict__ kbuf) {
    int idx = blockIdx.x * blockDim.x + threadIdx.x;  // over BH * T * (Dh/2)
    if (idx >= (B_N * H_N) << 16) return;
    const int i  = idx & 31;
    const int t  = (idx >> 5) & 2047;
    const int bh = idx >> 16;
    // inv_freq = 10000^(-2i/64) = exp(-(2i/64) * ln(10000))
    const float ang = (float)t * __expf(-(float)(2 * i) * (9.210340371976184f / 64.0f));
    float s, c;
    __sincosf(ang, &s, &c);
    const size_t base = ((size_t)bh * T_SEQ + t) * D_H;
    {
        float x1 = (float)qbuf[base + i], x2 = (float)qbuf[base + i + 32];
        qbuf[base + i]      = (bf16_t)(x1 * c - x2 * s);
        qbuf[base + i + 32] = (bf16_t)(x2 * c + x1 * s);
    }
    {
        float x1 = (float)kbuf[base + i], x2 = (float)kbuf[base + i + 32];
        kbuf[base + i]      = (bf16_t)(x1 * c - x2 * s);
        kbuf[base + i + 32] = (bf16_t)(x2 * c + x1 * s);
    }
}

// ---------------- stage 3: flash attention ----------------
// block: 128 threads = 4 waves; each wave owns 16 query rows. grid = (T/64, B*H).
__global__ __launch_bounds__(128) void attn_kernel(
    const bf16_t* __restrict__ qbuf,  // [BH][T][Dh]
    const bf16_t* __restrict__ kbuf,  // [BH][T][Dh]
    const bf16_t* __restrict__ vbuf,  // [BH][Dh][T]
    bf16_t* __restrict__ obuf)        // [8192][1024] (m = b*T+t, col = h*64+d)
{
    __shared__ __attribute__((aligned(16))) bf16_t Ps[4 * 16 * 64];  // per-wave P tiles
    const int tid  = threadIdx.x;
    const int lane = tid & 31, wv = tid >> 5;
    const int half = lane >> 4, ln = lane & 15;
    const int bh  = blockIdx.y;
    const int qt0 = blockIdx.x * 64;
    const size_t qkbase = (size_t)bh * T_SEQ * D_H;
    const size_t vbase  = (size_t)bh * D_H * T_SEQ;

    // Q fragments stay resident (A-layout: row = lane%16, contiguous d)
    const int tq = qt0 + wv * 16 + ln;
    v16bf aq[2];
#pragma unroll
    for (int c = 0; c < 2; ++c)
        aq[c] = ld_frag(&qbuf[qkbase + (size_t)tq * D_H + c * 32 + half * 8]);

    float mrun[8], lrun[8];
    v8f acc[4] = {};
#pragma unroll
    for (int r = 0; r < 8; ++r) { mrun[r] = -1e30f; lrun[r] = 0.0f; }

    for (int kt = 0; kt < T_SEQ / 64; ++kt) {
        const int tk0 = kt * 64;
        // S = (Q @ K^T) * scale
        v8f s[4];
#pragma unroll
        for (int j = 0; j < 4; ++j) {
            v8f sj = {};
#pragma unroll
            for (int c = 0; c < 2; ++c) {
                v16bf kf = ld_frag(&kbuf[qkbase + (size_t)(tk0 + j * 16 + ln) * D_H + c * 32 + half * 8]);
                sj = wmma_bf16(aq[c], kf, sj);
            }
#pragma unroll
            for (int r = 0; r < 8; ++r) sj[r] *= 0.125f;   // 1/sqrt(64)
            s[j] = sj;
        }
        // row max (rows 0-7 live in lanes 0-15, rows 8-15 in lanes 16-31; xor<=8 stays in half)
        float rmax[8];
#pragma unroll
        for (int r = 0; r < 8; ++r)
            rmax[r] = fmaxf(fmaxf(s[0][r], s[1][r]), fmaxf(s[2][r], s[3][r]));
#pragma unroll
        for (int off = 1; off < 16; off <<= 1)
#pragma unroll
            for (int r = 0; r < 8; ++r)
                rmax[r] = fmaxf(rmax[r], __shfl_xor(rmax[r], off, 32));

        float mn[8], sc[8], ps[8];
#pragma unroll
        for (int r = 0; r < 8; ++r) {
            mn[r] = fmaxf(mrun[r], rmax[r]);
            sc[r] = __expf(mrun[r] - mn[r]);
            ps[r] = 0.0f;
            mrun[r] = mn[r];
        }
        // P = exp(S - mn), stash bf16 into LDS (C-layout -> [row][key] for A-frag reload)
#pragma unroll
        for (int j = 0; j < 4; ++j)
#pragma unroll
            for (int r = 0; r < 8; ++r) {
                float p = __expf(s[j][r] - mn[r]);
                ps[r] += p;
                Ps[(wv * 16 + r + half * 8) * 64 + j * 16 + ln] = (bf16_t)p;
            }
#pragma unroll
        for (int off = 1; off < 16; off <<= 1)
#pragma unroll
            for (int r = 0; r < 8; ++r)
                ps[r] += __shfl_xor(ps[r], off, 32);
#pragma unroll
        for (int r = 0; r < 8; ++r) lrun[r] = lrun[r] * sc[r] + ps[r];
#pragma unroll
        for (int j2 = 0; j2 < 4; ++j2)
#pragma unroll
            for (int r = 0; r < 8; ++r) acc[j2][r] *= sc[r];

        __syncthreads();
        v16bf ap[2];
#pragma unroll
        for (int c = 0; c < 2; ++c)
            ap[c] = ld_frag(&Ps[(wv * 16 + ln) * 64 + c * 32 + half * 8]);
        // O += P @ V   (V B-frag: lane = d column, contiguous key from [Dh][T] layout)
#pragma unroll
        for (int j2 = 0; j2 < 4; ++j2)
#pragma unroll
            for (int c = 0; c < 2; ++c) {
                v16bf vf = ld_frag(&vbuf[vbase + (size_t)(j2 * 16 + ln) * T_SEQ + tk0 + c * 32 + half * 8]);
                acc[j2] = wmma_bf16(ap[c], vf, acc[j2]);
            }
        __syncthreads();
    }

    // normalize + write [m][h*64+d]
    const int b = bh >> 4, h = bh & 15;
#pragma unroll
    for (int j2 = 0; j2 < 4; ++j2)
#pragma unroll
        for (int r = 0; r < 8; ++r) {
            const int m   = b * T_SEQ + qt0 + wv * 16 + half * 8 + r;
            const int col = h * 64 + j2 * 16 + ln;
            obuf[(size_t)m * C_DIM + col] = (bf16_t)(acc[j2][r] / lrun[r]);
        }
}

// ---------------- stage 4: output projection ----------------
__global__ __launch_bounds__(256) void gemm_out_kernel(
    const bf16_t* __restrict__ A,    // [8192][1024] (attention output)
    const bf16_t* __restrict__ Bt,   // [1024][1024] = W_out^T
    const float*  __restrict__ bias, // [1024]
    float* __restrict__ out)         // [8192][1024]
{
    __shared__ __attribute__((aligned(16))) bf16_t As[2][128 * 32];
    __shared__ __attribute__((aligned(16))) bf16_t Bs[2][128 * 32];
    const int tid  = threadIdx.x;
    const int lane = tid & 31, wv = tid >> 5;
    const int wr = wv & 3, wc = wv >> 2;
    const int half = lane >> 4, ln = lane & 15;
    const int m0 = blockIdx.y * 128;
    const int n0 = blockIdx.x * 128;
    const unsigned int ldsA0 = (unsigned int)(uintptr_t)&As[0][0];
    const unsigned int ldsB0 = (unsigned int)(uintptr_t)&Bs[0][0];
    const unsigned int bufBytes = 128u * 32u * 2u;

    v8f acc[2][4] = {};
    const int KT = C_DIM / 32;
    stage_async(ldsA0, ldsB0, A, Bt, m0, n0, 0, tid);
    for (int kt = 0; kt < KT; ++kt) {
        const int cur = kt & 1;
        if (kt + 1 < KT) {
            stage_async(ldsA0 + (cur ^ 1) * bufBytes, ldsB0 + (cur ^ 1) * bufBytes,
                        A, Bt, m0, n0, (kt + 1) * 32, tid);
            ASYNC_WAIT_4();
        } else {
            ASYNC_WAIT_0();
        }
        __syncthreads();

        v16bf af[2], bfg[4];
#pragma unroll
        for (int mi = 0; mi < 2; ++mi)
            af[mi] = ld_frag(&As[cur][(wr * 32 + mi * 16 + ln) * 32 + half * 8]);
#pragma unroll
        for (int ni = 0; ni < 4; ++ni)
            bfg[ni] = ld_frag(&Bs[cur][(wc * 64 + ni * 16 + ln) * 32 + half * 8]);
#pragma unroll
        for (int mi = 0; mi < 2; ++mi)
#pragma unroll
            for (int ni = 0; ni < 4; ++ni)
                acc[mi][ni] = wmma_bf16(af[mi], bfg[ni], acc[mi][ni]);
        __syncthreads();
    }
#pragma unroll
    for (int mi = 0; mi < 2; ++mi)
#pragma unroll
        for (int ni = 0; ni < 4; ++ni) {
            const int gm = m0 + wr * 32 + mi * 16 + half * 8;
            const int gn = n0 + wc * 64 + ni * 16 + ln;
            const float bv = bias[gn];
#pragma unroll
            for (int r = 0; r < 8; ++r)
                out[(size_t)(gm + r) * C_DIM + gn] = acc[mi][ni][r] + bv;
        }
}

// ---------------- launcher ----------------
extern "C" void kernel_launch(void* const* d_in, const int* in_sizes, int n_in,
                              void* d_out, int out_size, void* d_ws, size_t ws_size,
                              hipStream_t stream) {
    const float* x     = (const float*)d_in[0];  // [4,2048,1024]
    const float* W_qkv = (const float*)d_in[1];  // [1024,3072]
    const float* b_qkv = (const float*)d_in[2];  // [3072]
    const float* W_out = (const float*)d_in[3];  // [1024,1024]
    const float* b_out = (const float*)d_in[4];  // [1024]
    float* out = (float*)d_out;

    char* ws = (char*)d_ws;
    bf16_t* xb    = (bf16_t*)(ws);                         // 16 MiB (reused as obuf)
    bf16_t* wqkvT = (bf16_t*)(ws + (size_t)16777216);      //  6 MiB
    bf16_t* woT   = (bf16_t*)(ws + (size_t)23068672);      //  2 MiB
    bf16_t* qb    = (bf16_t*)(ws + (size_t)25165824);      // 16 MiB
    bf16_t* kb    = (bf16_t*)(ws + (size_t)41943040);      // 16 MiB
    bf16_t* vb    = (bf16_t*)(ws + (size_t)58720256);      // 16 MiB -> 72 MiB total
    bf16_t* obuf  = xb;  // x is fully consumed by gemm_qkv before attn writes obuf

    // stage 0: precision conversion / weight transposition
    f32_to_bf16_kernel<<<(M_ROWS * C_DIM) / 256, 256, 0, stream>>>(x, xb, M_ROWS * C_DIM);
    transpose_bf16_kernel<<<(C_DIM * N_QKV) / 256, 256, 0, stream>>>(W_qkv, wqkvT, C_DIM, N_QKV);
    transpose_bf16_kernel<<<(C_DIM * C_DIM) / 256, 256, 0, stream>>>(W_out, woT, C_DIM, C_DIM);

    // stage 1: QKV projection (WMMA, async double-buffered LDS) + q/k/v split, v transposed
    gemm_qkv_kernel<<<dim3(N_QKV / 128, M_ROWS / 128), 256, 0, stream>>>(xb, wqkvT, b_qkv, qb, kb, vb);

    // stage 2: RoPE
    rope_kernel<<<((B_N * H_N) << 16) / 256, 256, 0, stream>>>(qb, kb);

    // stage 3: attention (WMMA flash-style)
    attn_kernel<<<dim3(T_SEQ / 64, B_N * H_N), 128, 0, stream>>>(qb, kb, vb, obuf);

    // stage 4: output projection (WMMA, async double-buffered LDS) + bias, fp32 out
    gemm_out_kernel<<<dim3(C_DIM / 128, M_ROWS / 128), 256, 0, stream>>>(obuf, woT, b_out, out);
}
